// GPN_89910845374931
// MI455X (gfx1250) — matrix operand order repaired and checked by reading, not compiled, not run on hardware
//
#include <hip/hip_runtime.h>
#include <hip/hip_bf16.h>

#define B_    128
#define S1_   4096
#define FEAT_ 5
#define H_    128
#define LDT   136   // padded bf16 row stride for LDS tiles (272B -> 4-bank row skew)

typedef __attribute__((ext_vector_type(16))) __bf16 bf16x16;
typedef __attribute__((ext_vector_type(8)))  float  floatx8;

struct Ptrs { const float* p[58]; };

// d_in layout (setup_inputs dict order, params dict in insertion order):
// 0 x, 1 X_all, 2 mask, 3 mask_fin, 4 h, 5 c, 6 latent,
// 7.. pairs (W,b) for names: emb_x(0) emb_all(1) W1(2) W2(3) W3(4) agg1(5) agg2(6)
//   agg3(7) Wq(8) Wref(9) Wxi(10) Whi(11) wci(12) Wxf(13) Whf(14) wcf(15)
//   Wxc(16) Whc(17) Wxo(18) Who(19) wco(20)
// 49 v, 50 xn_w, 51 xn_b, 52 cn_w, 53 cn_b, 54 alpha, 55 r1, 56 r2, 57 r3
#define IW(i) (7 + 2*(i))
#define IB(i) (8 + 2*(i))

union Frag { bf16x16 v; unsigned int u[8]; };

// A-matrix 16x32 bf16 fragment: lanes 0-15 -> K 0..15,
// lanes 16-31 -> K 16..31 (contiguous, matches B), pairs per VGPR.
__device__ __forceinline__ bf16x16 load_frag_a(const __bf16* base, int row0, int k0, int lane) {
  const int l15 = lane & 15, half = (lane >> 4) & 1;
  const __bf16* rp = base + (row0 + l15) * LDT + k0 + 16 * half;
  const unsigned int* p0 = (const unsigned int*)rp;        // 16B aligned
  const unsigned int* p1 = (const unsigned int*)(rp + 8);  // 16B aligned
  Frag f;
#pragma unroll
  for (int i = 0; i < 4; ++i) { f.u[i] = p0[i]; f.u[4 + i] = p1[i]; }
  return f.v;
}

// B-matrix 32x16 bf16 fragment: lane = N column; lanes 0-15 hold K 0..15,
// lanes 16-31 hold K 16..31 (contiguous 32B per lane). W stored row-major (N x K),
// so B[k][n] = W[n][k] is a contiguous run per lane.
__device__ __forceinline__ bf16x16 load_frag_b(const __bf16* base, int n0, int k0, int lane) {
  const int l15 = lane & 15, half = (lane >> 4) & 1;
  const unsigned int* p = (const unsigned int*)(base + (n0 + l15) * LDT + k0 + 16 * half);
  Frag f;
#pragma unroll
  for (int i = 0; i < 8; ++i) f.u[i] = p[i];
  return f.v;
}

__device__ __forceinline__ float dot128(const float* v, const float* Wrow) {
  float s = 0.f;
#pragma unroll 8
  for (int k = 0; k < H_; ++k) s += v[k] * Wrow[k];
  return s;
}

__device__ __forceinline__ float sigmoidf(float x) { return 1.0f / (1.0f + expf(-x)); }

// ---------------- kernel 0: zero LN accumulators ----------------
__global__ void k_init(float* sums, float* sqs) {
  int t = threadIdx.x;
  if (t < B_) { sums[t] = 0.f; sqs[t] = 0.f; }
}

// ---------------- kernel 1: xe-LN + gated LSTM + q ----------------
__global__ __launch_bounds__(128)
void k_lstm(Ptrs P, float* out_h, float* out_c, float* ws_q) {
  __shared__ float xs[FEAT_], xe[H_], hh[H_], cc[H_], cn[H_], hn[H_], red[H_];
  const int b = blockIdx.x, t = threadIdx.x;
  if (t < FEAT_) xs[t] = P.p[0][b * FEAT_ + t];
  hh[t] = P.p[4][b * H_ + t];
  cc[t] = P.p[5][b * H_ + t];
  __syncthreads();

  const float* exW = P.p[IW(0)];
  float e = P.p[IB(0)][t];
#pragma unroll
  for (int k = 0; k < FEAT_; ++k) e += xs[k] * exW[t * FEAT_ + k];

  // LayerNorm over hidden dim
  red[t] = e; __syncthreads();
  for (int off = 64; off > 0; off >>= 1) { if (t < off) red[t] += red[t + off]; __syncthreads(); }
  const float mean = red[0] * (1.0f / H_);
  __syncthreads();
  const float d = e - mean;
  red[t] = d * d; __syncthreads();
  for (int off = 64; off > 0; off >>= 1) { if (t < off) red[t] += red[t + off]; __syncthreads(); }
  const float rstd = rsqrtf(red[0] * (1.0f / H_) + 1e-5f);
  xe[t] = d * rstd * P.p[50][t] + P.p[51][t];
  __syncthreads();

  float ig = sigmoidf(dot128(xe, P.p[IW(10)] + t * H_) + P.p[IB(10)][t] +
                      dot128(hh, P.p[IW(11)] + t * H_) + P.p[IB(11)][t] +
                      dot128(cc, P.p[IW(12)] + t * H_) + P.p[IB(12)][t]);
  float fg = sigmoidf(dot128(xe, P.p[IW(13)] + t * H_) + P.p[IB(13)][t] +
                      dot128(hh, P.p[IW(14)] + t * H_) + P.p[IB(14)][t] +
                      dot128(cc, P.p[IW(15)] + t * H_) + P.p[IB(15)][t]);
  float g  = tanhf(dot128(xe, P.p[IW(16)] + t * H_) + P.p[IB(16)][t] +
                   dot128(hh, P.p[IW(17)] + t * H_) + P.p[IB(17)][t]);
  float cnew = fg * cc[t] + ig * g;
  cn[t] = cnew; out_c[b * H_ + t] = cnew;
  __syncthreads();
  float og = sigmoidf(dot128(xe, P.p[IW(18)] + t * H_) + P.p[IB(18)][t] +
                      dot128(hh, P.p[IW(19)] + t * H_) + P.p[IB(19)][t] +
                      dot128(cn, P.p[IW(20)] + t * H_) + P.p[IB(20)][t]);
  float hnew = og * tanhf(cnew);
  hn[t] = hnew; out_h[b * H_ + t] = hnew;
  __syncthreads();
  ws_q[b * H_ + t] = dot128(hn, P.p[IW(8)] + t * H_) + P.p[IB(8)][t];
}

// ---------------- kernel 2: per-batch embed sum/sumsq ----------------
__global__ __launch_bounds__(256)
void k_stats(Ptrs P, float* sums, float* sqs) {
  const int blk = blockIdx.x;
  const int b = blk >> 8;            // 256 chunks of 16 rows
  const int s0 = (blk & 255) * 16;
  const int t = threadIdx.x;
  __shared__ float Xs[16 * FEAT_];
  __shared__ float Ws[H_ * FEAT_ + H_];
  __shared__ float red[256];
  const float* Xall = P.p[1];
  for (int i = t; i < 16 * FEAT_; i += 256)
    Xs[i] = Xall[((size_t)b * S1_ + s0) * FEAT_ + i];
  const float* eW = P.p[IW(1)];
  for (int i = t; i < H_ * FEAT_; i += 256) Ws[i] = eW[i];
  for (int i = t; i < H_; i += 256) Ws[H_ * FEAT_ + i] = P.p[IB(1)][i];
  __syncthreads();

  float sum = 0.f, sq = 0.f;
#pragma unroll
  for (int i = 0; i < 8; ++i) {
    int flat = i * 256 + t, m = flat >> 7, j = flat & 127;
    float e = Ws[H_ * FEAT_ + j];
#pragma unroll
    for (int k = 0; k < FEAT_; ++k) e += Xs[m * FEAT_ + k] * Ws[j * FEAT_ + k];
    sum += e; sq += e * e;
  }
  red[t] = sum; __syncthreads();
  for (int off = 128; off > 0; off >>= 1) { if (t < off) red[t] += red[t + off]; __syncthreads(); }
  if (t == 0) atomicAdd(&sums[b], red[0]);
  __syncthreads();
  red[t] = sq; __syncthreads();
  for (int off = 128; off > 0; off >>= 1) { if (t < off) red[t] += red[t + off]; __syncthreads(); }
  if (t == 0) atomicAdd(&sqs[b], red[0]);
}

// ---------------- kernel 3: fused LN + 3 GNN stages + Wref + pointer ----------------
__global__ __launch_bounds__(256)
void k_main(Ptrs P, const float* sums, const float* sqs, const float* ws_q, float* out_u) {
  __shared__ __bf16 ctx[H_ * LDT];     // activations, updated in place
  __shared__ __bf16 wW[H_ * LDT];      // stage weight 1 (bf16)
  __shared__ __bf16 wA[H_ * LDT];      // stage weight 2 (bf16)
  __shared__ float biasW[H_], biasA[H_];
  __shared__ float qpb[H_], vv[H_];
  __shared__ float Xt[H_ * FEAT_];
  __shared__ float eWb[H_ * FEAT_ + H_];

  const int blk = blockIdx.x;
  const int b = blk >> 5;
  const int s0 = (blk & 31) * H_;
  const int t = threadIdx.x;
  const int lane = t & 31;
  const int wv = t >> 5;               // 8 waves, each owns 16 ctx rows
  const int m0 = wv * 16;
  const int l15 = lane & 15, half = (lane >> 4) & 1;

  const float invN = 1.0f / (float)(S1_ * H_);
  const float mean = sums[b] * invN;
  const float rstd = rsqrtf(sqs[b] * invN - mean * mean + 1e-5f);

  const float* Xall = P.p[1];
  for (int i = t; i < H_ * FEAT_; i += 256)
    Xt[i] = Xall[((size_t)b * S1_ + s0) * FEAT_ + i];
  const float* eW = P.p[IW(1)];
  for (int i = t; i < H_ * FEAT_; i += 256) eWb[i] = eW[i];
  for (int i = t; i < H_; i += 256) eWb[H_ * FEAT_ + i] = P.p[IB(1)][i];
  __syncthreads();

  // recompute embedding + joint LayerNorm -> bf16 ctx tile
  const float* cnw = P.p[52]; const float* cnb = P.p[53];
  for (int i = t; i < H_ * H_; i += 256) {
    int m = i >> 7, j = i & 127;
    float e = eWb[H_ * FEAT_ + j];
#pragma unroll
    for (int k = 0; k < FEAT_; ++k) e += Xt[m * FEAT_ + k] * eWb[j * FEAT_ + k];
    size_t g = (size_t)(s0 + m) * H_ + j;
    ctx[m * LDT + j] = (__bf16)((e - mean) * rstd * cnw[g] + cnb[g]);
  }
  __syncthreads();

  const float inv_n = 1.0f / (float)(S1_ - 1);

  // ---- 3 GNN stages, in place (results ride in accumulators across barrier)
  for (int st = 0; st < 3; ++st) {
    const float* Wt = P.p[IW(2 + st)];
    const float* At = P.p[IW(5 + st)];
    for (int i = t; i < H_ * H_; i += 256) {
      int m = i >> 7, j = i & 127;
      wW[m * LDT + j] = (__bf16)Wt[i];
      wA[m * LDT + j] = (__bf16)At[i];
    }
    for (int i = t; i < H_; i += 256) { biasW[i] = P.p[IB(2 + st)][i]; biasA[i] = P.p[IB(5 + st)][i]; }
    const float rr = P.p[55 + st][0];
    __syncthreads();

    floatx8 acc1[8], acc2[8];
#pragma unroll
    for (int n = 0; n < 8; ++n) { acc1[n] = (floatx8){0,0,0,0,0,0,0,0}; acc2[n] = (floatx8){0,0,0,0,0,0,0,0}; }

    // software-pipelined: B frags for tile n+1 and A frag for k+1 are issued
    // before the WMMAs of tile n, so ds loads overlap the matrix pipe.
    bf16x16 a_cur = load_frag_a(ctx, m0, 0, lane);
#pragma unroll
    for (int kk = 0; kk < 4; ++kk) {
      bf16x16 a_next = (kk < 3) ? load_frag_a(ctx, m0, (kk + 1) * 32, lane) : a_cur;
      bf16x16 bw = load_frag_b(wW, 0, kk * 32, lane);
      bf16x16 ba = load_frag_b(wA, 0, kk * 32, lane);
#pragma unroll
      for (int n = 0; n < 8; ++n) {
        bf16x16 bw_n = bw, ba_n = ba;
        if (n < 7) {
          bw_n = load_frag_b(wW, (n + 1) * 16, kk * 32, lane);
          ba_n = load_frag_b(wA, (n + 1) * 16, kk * 32, lane);
        }
        acc1[n] = __builtin_amdgcn_wmma_f32_16x16x32_bf16(false, a_cur, false, bw, (short)0, acc1[n], false, false);
        acc2[n] = __builtin_amdgcn_wmma_f32_16x16x32_bf16(false, a_cur, false, ba, (short)0, acc2[n], false, false);
        bw = bw_n; ba = ba_n;
      }
      a_cur = a_next;
    }
    __syncthreads();   // all waves done reading ctx -> safe to overwrite in place
#pragma unroll
    for (int n = 0; n < 8; ++n) {
      int col = n * 16 + l15;
      float bwv = biasW[col], bav = biasA[col];
#pragma unroll
      for (int r = 0; r < 8; ++r) {
        int row = m0 + r + 8 * half;
        float y1 = acc1[n][r] + bwv;
        float y2 = inv_n * acc2[n][r] + bav;
        ctx[row * LDT + col] = (__bf16)(rr * y1 + (1.0f - rr) * fmaxf(y2, 0.0f));
      }
    }
    __syncthreads();
  }

  // ---- Wref GEMM + tanh pointer reduction
  {
    const float* Wt = P.p[IW(9)]; const float* bt = P.p[IB(9)];
    for (int i = t; i < H_ * H_; i += 256) {
      int m = i >> 7, j = i & 127;
      wW[m * LDT + j] = (__bf16)Wt[i];
    }
    const float* q = ws_q + b * H_;
    const float* vp = P.p[49];
    for (int i = t; i < H_; i += 256) { qpb[i] = q[i] + bt[i]; vv[i] = vp[i]; }
    __syncthreads();

    floatx8 acc[8];
#pragma unroll
    for (int n = 0; n < 8; ++n) acc[n] = (floatx8){0,0,0,0,0,0,0,0};

    bf16x16 a_cur = load_frag_a(ctx, m0, 0, lane);
#pragma unroll
    for (int kk = 0; kk < 4; ++kk) {
      bf16x16 a_next = (kk < 3) ? load_frag_a(ctx, m0, (kk + 1) * 32, lane) : a_cur;
      bf16x16 bw = load_frag_b(wW, 0, kk * 32, lane);
#pragma unroll
      for (int n = 0; n < 8; ++n) {
        bf16x16 bw_n = bw;
        if (n < 7) bw_n = load_frag_b(wW, (n + 1) * 16, kk * 32, lane);
        acc[n] = __builtin_amdgcn_wmma_f32_16x16x32_bf16(false, a_cur, false, bw, (short)0, acc[n], false, false);
        bw = bw_n;
      }
      a_cur = a_next;
    }

    float rs[8] = {0, 0, 0, 0, 0, 0, 0, 0};
#pragma unroll
    for (int n = 0; n < 8; ++n) {
      int col = n * 16 + l15;
      float qv = qpb[col], vw = vv[col];
#pragma unroll
      for (int r = 0; r < 8; ++r) rs[r] += vw * tanhf(qv + acc[n][r]);
    }
#pragma unroll
    for (int r = 0; r < 8; ++r) {      // reduce within 16-lane halves (C layout)
      float x = rs[r];
      x += __shfl_xor(x, 1, 32);
      x += __shfl_xor(x, 2, 32);
      x += __shfl_xor(x, 4, 32);
      x += __shfl_xor(x, 8, 32);
      rs[r] = x;
    }
    const float alpha = P.p[54][0];
    const float* mask = P.p[2]; const float* lat = P.p[6];
    if (l15 == 0) {
#pragma unroll
      for (int r = 0; r < 8; ++r) {
        size_t g = (size_t)b * S1_ + s0 + m0 + r + 8 * half;
        out_u[g] = 10.0f * tanhf(rs[r]) + mask[g] + alpha * lat[g];
      }
    }
  }
}

// ---------------- kernel 4: row softmax ----------------
__global__ __launch_bounds__(256)
void k_softmax(const float* u, float* probs) {
  const int b = blockIdx.x, t = threadIdx.x;
  __shared__ float red[256];
  const float* ub = u + (size_t)b * S1_;
  float mx = -3.4e38f;
  for (int i = t; i < S1_; i += 256) mx = fmaxf(mx, ub[i]);
  red[t] = mx; __syncthreads();
  for (int off = 128; off > 0; off >>= 1) { if (t < off) red[t] = fmaxf(red[t], red[t + off]); __syncthreads(); }
  mx = red[0]; __syncthreads();
  float s = 0.f;
  for (int i = t; i < S1_; i += 256) s += expf(ub[i] - mx);
  red[t] = s; __syncthreads();
  for (int off = 128; off > 0; off >>= 1) { if (t < off) red[t] += red[t + off]; __syncthreads(); }
  const float inv = 1.0f / red[0];
  for (int i = t; i < S1_; i += 256) probs[(size_t)b * S1_ + i] = expf(ub[i] - mx) * inv;
}

extern "C" void kernel_launch(void* const* d_in, const int* in_sizes, int n_in,
                              void* d_out, int out_size, void* d_ws, size_t ws_size,
                              hipStream_t stream) {
  Ptrs P;
  for (int i = 0; i < 58 && i < n_in; ++i) P.p[i] = (const float*)d_in[i];

  float* ws   = (float*)d_ws;
  float* sums = ws;             // B
  float* sqs  = ws + 128;       // B
  float* ws_q = ws + 256;       // B*H

  float* out       = (float*)d_out;       // (probs, h_new, c_new, u) concatenated
  float* out_probs = out;
  float* out_h     = out + (size_t)B_ * S1_;
  float* out_c     = out_h + (size_t)B_ * H_;
  float* out_u     = out_c + (size_t)B_ * H_;

  k_init   <<<1, 128, 0, stream>>>(sums, sqs);
  k_lstm   <<<B_, 128, 0, stream>>>(P, out_h, out_c, ws_q);
  k_stats  <<<B_ * (S1_ / 16), 256, 0, stream>>>(P, sums, sqs);
  k_main   <<<B_ * (S1_ / 128), 256, 0, stream>>>(P, sums, sqs, ws_q, out_u);
  k_softmax<<<B_, 256, 0, stream>>>(out_u, out_probs);
}
